// EuclideanCodebook_88510686036498
// MI455X (gfx1250) — matrix operand order repaired and checked by reading, not compiled, not run on hardware
//
#include <hip/hip_runtime.h>
#include <cstdint>

typedef __attribute__((ext_vector_type(16))) __bf16 v16bf;
typedef __attribute__((ext_vector_type(8)))  float  v8f;
typedef unsigned short u16;

#define DIMC 256
#define KCODES 8192
#define NROWS 32768
#define TILE_M 256          // rows per block (8 waves * 32)
#define NTILES (KCODES / 16)

union Frag {
    uint4 u[2];
    v16bf v;
};

__device__ __forceinline__ u16 f2bf(float f) {
    unsigned u = __float_as_uint(f);
    unsigned r = u + 0x7FFFu + ((u >> 16) & 1u);   // round-to-nearest-even
    return (u16)(r >> 16);
}

// ---------------- prep: fp32 -> bf16 (vectorized x4) ----------------
__global__ void k_prep_bf16(const float* __restrict__ in, u16* __restrict__ out, int n4) {
    int i = blockIdx.x * blockDim.x + threadIdx.x;
    if (i >= n4) return;
    float4 f = ((const float4*)in)[i];
    ushort4 o;
    o.x = f2bf(f.x); o.y = f2bf(f.y); o.z = f2bf(f.z); o.w = f2bf(f.w);
    ((ushort4*)out)[i] = o;
}

// ---------------- prep: |e|^2 per codebook row ----------------
__global__ void k_esq(const float* __restrict__ e, float* __restrict__ esq) {
    int r = blockIdx.x * blockDim.x + threadIdx.x;   // 8192 rows
    const float4* p = (const float4*)(e + (size_t)r * DIMC);
    float s = 0.f;
#pragma unroll 8
    for (int i = 0; i < DIMC / 4; ++i) {
        float4 v = p[i];
        s += v.x * v.x + v.y * v.y + v.z * v.z + v.w * v.w;
    }
    esq[r] = s;
}

// ---------------- main: fused WMMA GEMM + running argmax ----------------
// Block: 256 threads = 8 waves. Wave w owns 32 rows: blockIdx.x*256 + w*32 .. +31
// (two 16-row M-tiles sharing every B fragment -> 16 WMMAs per 16 ds-load pair,
//  two independent accumulation chains for XDL ILP).
// Loop over 512 codebook tiles of 16 entries; tile staged in LDS (double
// buffered) in WMMA-B fragment order.
__global__ __launch_bounds__(256) void
k_argmax(const u16* __restrict__ xb, const u16* __restrict__ eb,
         const float* __restrict__ esq, int* __restrict__ outIdx) {
    __shared__ alignas(16) unsigned char sraw[32768];   // 2x8KB B buffers; 32KB argmax scratch

    const int tid  = threadIdx.x;
    const int wave = tid >> 5;
    const int lane = tid & 31;
    const int lmod = lane & 15;
    const int lsel = lane >> 4;
    const int row0 = blockIdx.x * TILE_M + wave * 32 + lmod;   // M-tile 0
    const int row1 = row0 + 16;                                // M-tile 1

    // ---- load A fragments once: 2 x (16 rows x 256 C) bf16 per wave ----
    // A frags in contiguous k order (B staged to match): lanes 0-15 hold K {c0..c0+15},
    //                 lanes 16-31 hold K {c0+16..c0+31}.
    v16bf afrag0[8], afrag1[8];
    {
        const uint4* p0 = (const uint4*)(xb + (size_t)row0 * DIMC);
        const uint4* p1 = (const uint4*)(xb + (size_t)row1 * DIMC);
#pragma unroll
        for (int c = 0; c < 8; ++c) {
            int bi = (c * 32 + lsel * 16) >> 3;   // uint4 index (8 bf16 per uint4)
            Frag fa;
            fa.u[0] = p0[bi];
            fa.u[1] = p0[bi + 1];                // +8 elements
            afrag0[c] = fa.v;
            Frag fb;
            fb.u[0] = p1[bi];
            fb.u[1] = p1[bi + 1];
            afrag1[c] = fb.v;
        }
    }

    unsigned char* buf0 = sraw;
    unsigned char* buf1 = sraw + 8192;

    // ---- cooperative stage of one 16-entry x 256-C bf16 tile into LDS ----
    // slot t (0..255): chunk c = t>>5, lane l = t&31 ->
    //   16 bf16 = embed[n0 + (l&15)][c*32 + (l>=16)*16 ...], stored contiguously.
    auto stage = [&](unsigned char* dst, int n0) {
        int c = tid >> 5;
        int l = tid & 31;
        int n = n0 + (l & 15);
        int coff = c * 32 + ((l >> 4) << 4);
        const uint4* src = (const uint4*)(eb + (size_t)n * DIMC + coff);
        uint4 v0 = src[0];
        uint4 v1 = src[1];
        uint4* d = (uint4*)(dst + tid * 32);
        d[0] = v0;
        d[1] = v1;
    };

    stage(buf0, 0);

    float bestS0[8], bestS1[8];
    int   bestI0[8], bestI1[8];
#pragma unroll
    for (int r = 0; r < 8; ++r) {
        bestS0[r] = -__builtin_inff(); bestI0[r] = 0;
        bestS1[r] = -__builtin_inff(); bestI1[r] = 0;
    }

    for (int it = 0; it < NTILES; ++it) {
        const int n0 = it << 4;
        __syncthreads();
        unsigned char* cur = (it & 1) ? buf1 : buf0;
        if (it + 1 < NTILES) {
            if (it + 2 < NTILES) {
                // speculative prefetch of the tile two steps ahead
                int pn = (it + 2) * 16 + ((tid & 31) & 15);
                int pc = (tid >> 5) * 32 + ((((tid & 31)) >> 4) << 4);
                __builtin_prefetch(eb + (size_t)pn * DIMC + pc, 0, 1);
            }
            stage((it & 1) ? buf0 : buf1, n0 + 16);
        }

        float eq = esq[n0 + lmod];

        v8f acc0 = {};
        v8f acc1 = {};
#pragma unroll
        for (int c = 0; c < 8; ++c) {
            Frag fb;
            const uint4* p = (const uint4*)(cur + ((c * 32 + lane) << 5));
            fb.u[0] = p[0];
            fb.u[1] = p[1];
            acc0 = __builtin_amdgcn_wmma_f32_16x16x32_bf16(
                false, afrag0[c], false, fb.v, (short)0, acc0, false, false);
            acc1 = __builtin_amdgcn_wmma_f32_16x16x32_bf16(
                false, afrag1[c], false, fb.v, (short)0, acc1, false, false);
        }

        // score = 2*dot - |e|^2 ; per-slot running argmax
        const int col = n0 + lmod;
#pragma unroll
        for (int r = 0; r < 8; ++r) {
            float s0 = 2.0f * acc0[r] - eq;
            if (s0 > bestS0[r]) { bestS0[r] = s0; bestI0[r] = col; }
            float s1 = 2.0f * acc1[r] - eq;
            if (s1 > bestS1[r]) { bestS1[r] = s1; bestI1[r] = col; }
        }
    }

    // ---- cross-lane argmax reduction (reuse all 32KB of LDS) ----
    __syncthreads();
    float* rs = (float*)sraw;                        // [wave][lane][tile*8+r] scores, 16KB
    int*   ri = (int*)(sraw + 16384);                // indices, 16KB
#pragma unroll
    for (int r = 0; r < 8; ++r) {
        rs[(wave * 32 + lane) * 16 + r]     = bestS0[r];
        ri[(wave * 32 + lane) * 16 + r]     = bestI0[r];
        rs[(wave * 32 + lane) * 16 + 8 + r] = bestS1[r];
        ri[(wave * 32 + lane) * 16 + 8 + r] = bestI1[r];
    }
    __syncthreads();
    {
        int m    = tid;                 // local row 0..255
        int w    = m >> 5;
        int mr   = m & 31;
        int t    = mr >> 4;             // which M-tile
        int rr   = mr & 15;
        int half = rr >> 3;             // row lives in lanes [half*16, half*16+16)
        int r    = rr & 7;
        float bs = -__builtin_inff();
        int   bi = 0x7FFFFFFF;
        for (int l = half * 16; l < half * 16 + 16; ++l) {
            float s = rs[(w * 32 + l) * 16 + t * 8 + r];
            int   i = ri[(w * 32 + l) * 16 + t * 8 + r];
            if (s > bs || (s == bs && i < bi)) { bs = s; bi = i; }
        }
        outIdx[blockIdx.x * TILE_M + m] = bi;
    }
}

// ---------------- gather winning fp32 codebook rows ----------------
__global__ void k_gather(const float* __restrict__ embed, const int* __restrict__ idx,
                         float* __restrict__ out) {
    int row = blockIdx.x;               // 32768 rows
    int id  = idx[row];
    const float4* src = (const float4*)(embed + (size_t)id * DIMC);
    float4*       dst = (float4*)(out + (size_t)row * DIMC);
    dst[threadIdx.x] = src[threadIdx.x];   // 64 threads * float4 = 256 floats
}

extern "C" void kernel_launch(void* const* d_in, const int* in_sizes, int n_in,
                              void* d_out, int out_size, void* d_ws, size_t ws_size,
                              hipStream_t stream) {
    const float* x     = (const float*)d_in[0];   // (8,4096,256) fp32
    const float* embed = (const float*)d_in[1];   // (8192,256) fp32
    float* out = (float*)d_out;                   // (8,4096,256) fp32

    char* ws = (char*)d_ws;
    const size_t XB_BYTES = (size_t)NROWS * DIMC * sizeof(u16);   // 16 MB
    const size_t EB_BYTES = (size_t)KCODES * DIMC * sizeof(u16);  // 4 MB
    u16*   xb  = (u16*)ws;
    u16*   eb  = (u16*)(ws + XB_BYTES);
    float* esq = (float*)(ws + XB_BYTES + EB_BYTES);
    int*   idx = (int*)(ws + XB_BYTES + EB_BYTES + KCODES * sizeof(float));

    // convert x and embed to bf16
    {
        int n4 = NROWS * DIMC / 4;
        k_prep_bf16<<<n4 / 256, 256, 0, stream>>>(x, xb, n4);
    }
    {
        int n4 = KCODES * DIMC / 4;
        k_prep_bf16<<<n4 / 256, 256, 0, stream>>>(embed, eb, n4);
    }
    // |e|^2
    k_esq<<<KCODES / 256, 256, 0, stream>>>(embed, esq);

    // fused WMMA GEMM + argmax
    k_argmax<<<NROWS / TILE_M, 256, 0, stream>>>(xb, eb, esq, idx);

    // gather fp32 rows
    k_gather<<<NROWS, 64, 0, stream>>>(embed, idx, out);
}
